// GCN_21560735826552
// MI455X (gfx1250) — matrix operand order repaired and checked by reading, not compiled, not run on hardware
//
#include <hip/hip_runtime.h>

#define NFEAT  500
#define NHID   64
#define NCLASS 40
#define KT1    16   // ceil(500/32)
#define NT1    4    // 64/16
#define KT2    2    // 64/32
#define NT2    3    // ceil(40/16)

typedef __attribute__((ext_vector_type(16))) __bf16 v16bf;
typedef __attribute__((ext_vector_type(8)))  __bf16 v8bf;
typedef __attribute__((ext_vector_type(8)))  float  v8f;

union AFrag { v16bf v; v8bf h[2]; __bf16 e[16]; };

// ---------------------------------------------------------------------------
// Prep: swizzle a row-major [K x Ncols] fp32 weight matrix into per-wave
// bf16 B-fragments for V_WMMA_F32_16X16X32_BF16.
// Fragment element index t = ((kt*NT + nt)*32 + lane)*16 + i
// B 32x16 layout: lanes 0-15 hold K=[0..7],[16..23], lanes 16-31 hold
// K=[8..15],[24..31]; n = nt*16 + (lane&15).
// ---------------------------------------------------------------------------
__global__ __launch_bounds__(256) void prep_wfrag(const float* __restrict__ W,
                                                  __bf16* __restrict__ frag,
                                                  int K, int Ncols, int NT, int KT) {
  int t = blockIdx.x * blockDim.x + threadIdx.x;
  int total = KT * NT * 32 * 16;
  if (t >= total) return;
  int i    = t & 15;
  int lane = (t >> 4) & 31;
  int ntkt = t >> 9;
  int nt   = ntkt % NT;
  int kt   = ntkt / NT;
  int k = kt * 32 + ((i < 8) ? i : (8 + i)) + ((lane < 16) ? 0 : 8);
  int n = nt * 16 + (lane & 15);
  float v = (k < K && n < Ncols) ? W[(size_t)k * Ncols + n] : 0.0f;
  frag[t] = (__bf16)v;
}

__global__ __launch_bounds__(256) void zero_f32(float* __restrict__ p, long long n) {
  long long i = (long long)blockIdx.x * blockDim.x + threadIdx.x;
  if (i < n) p[i] = 0.0f;
}

// Load 8 consecutive fp32 (k0 multiple of 8 -> 32B aligned), convert to bf16.
__device__ __forceinline__ void load8_f32_bf16(const float* __restrict__ row,
                                               int k0, int K, __bf16* dst) {
  if (k0 + 8 <= K) {
    const float4* p = (const float4*)(row + k0);
    float4 q0 = p[0], q1 = p[1];
    dst[0]=(__bf16)q0.x; dst[1]=(__bf16)q0.y; dst[2]=(__bf16)q0.z; dst[3]=(__bf16)q0.w;
    dst[4]=(__bf16)q1.x; dst[5]=(__bf16)q1.y; dst[6]=(__bf16)q1.z; dst[7]=(__bf16)q1.w;
  } else {
#pragma unroll
    for (int j = 0; j < 8; ++j) {
      int k = k0 + j;
      dst[j] = (__bf16)((k < K) ? row[k] : 0.0f);
    }
  }
}

// ---------------------------------------------------------------------------
// GEMM1: sup[N][64] (bf16) = x[N][500] (fp32->bf16) @ W1frag
// One wave per 16-row tile, 4 waves per block. K padded to 512.
// A 16x32 bf16 layout: lanes 0-15 -> M=lane, chunks K=[0..7] and [16..23];
// lanes 16-31 -> M=lane-16, chunks K=[8..15] and [24..31].
// ---------------------------------------------------------------------------
__global__ __launch_bounds__(128) void gemm1_wmma(const float* __restrict__ x,
                                                  const __bf16* __restrict__ wfrag,
                                                  __bf16* __restrict__ sup,
                                                  int Ntiles, int N) {
  int wave = threadIdx.x >> 5;
  int lane = threadIdx.x & 31;
  int tile = blockIdx.x * 4 + wave;
  if (tile >= Ntiles) return;
  bool hi = lane >= 16;
  int m = tile * 16 + (lane & 15);
  int mc = m < N ? m : (N - 1);
  const float* xrow = x + (size_t)mc * NFEAT;
  const v16bf* bp = (const v16bf*)wfrag;

  v8f a0 = {}, a1 = {}, a2 = {}, a3 = {};
  for (int kt = 0; kt < KT1; ++kt) {
    AFrag af;
    int c0 = kt * 32 + (hi ? 8 : 0);
    int c1 = kt * 32 + (hi ? 24 : 16);
    load8_f32_bf16(xrow, c0, NFEAT, af.e);
    load8_f32_bf16(xrow, c1, NFEAT, af.e + 8);
    v16bf fb0 = bp[(kt * NT1 + 0) * 32 + lane];
    v16bf fb1 = bp[(kt * NT1 + 1) * 32 + lane];
    v16bf fb2 = bp[(kt * NT1 + 2) * 32 + lane];
    v16bf fb3 = bp[(kt * NT1 + 3) * 32 + lane];
    a0 = __builtin_amdgcn_wmma_f32_16x16x32_bf16(false, af.v, false, fb0, (short)0, a0, false, false);
    a1 = __builtin_amdgcn_wmma_f32_16x16x32_bf16(false, af.v, false, fb1, (short)0, a1, false, false);
    a2 = __builtin_amdgcn_wmma_f32_16x16x32_bf16(false, af.v, false, fb2, (short)0, a2, false, false);
    a3 = __builtin_amdgcn_wmma_f32_16x16x32_bf16(false, af.v, false, fb3, (short)0, a3, false, false);
  }
  // C layout: VGPR r -> (M = r + (hi?8:0), N = lane&15)
  int mbase = tile * 16 + (hi ? 8 : 0);
  int nn = lane & 15;
#pragma unroll
  for (int r = 0; r < 8; ++r) {
    int mm = mbase + r;
    if (mm < N) {
      __bf16* orow = sup + (size_t)mm * NHID;
      orow[ 0 + nn] = (__bf16)a0[r];
      orow[16 + nn] = (__bf16)a1[r];
      orow[32 + nn] = (__bf16)a2[r];
      orow[48 + nn] = (__bf16)a3[r];
    }
  }
}

// ---------------------------------------------------------------------------
// Edge aggregation: agg[dst] += sup[src] * w, fp32 atomics, bf16 gather.
// CH = D/4 chunks of 4 bf16 (8-byte aligned) per edge.
// ---------------------------------------------------------------------------
__global__ __launch_bounds__(256) void aggregate_bf16(const int* __restrict__ srcv,
                                                      const int* __restrict__ dstv,
                                                      const float* __restrict__ ew,
                                                      const __bf16* __restrict__ sup,
                                                      float* __restrict__ agg,
                                                      int E, int D, int CH) {
  long long tid = (long long)blockIdx.x * blockDim.x + threadIdx.x;
  int e = (int)(tid / CH);
  int c = (int)(tid % CH) * 4;
  if (e >= E) return;
  int s = srcv[e], d = dstv[e];
  float w = ew[e];
  unsigned long long bits = *(const unsigned long long*)(sup + (size_t)s * D + c);
  float* out = agg + (size_t)d * D + c;
#pragma unroll
  for (int j = 0; j < 4; ++j) {
    unsigned u = (unsigned)((bits >> (16 * j)) & 0xffffu) << 16;
    atomicAdd(out + j, __uint_as_float(u) * w);
  }
}

// h = relu(agg1 + b1) * drop_mask, stored bf16 (reuses sup1 buffer)
__global__ __launch_bounds__(256) void post1_relu_drop(const float* __restrict__ agg,
                                                       const float* __restrict__ b1,
                                                       const float* __restrict__ mask,
                                                       __bf16* __restrict__ hbuf,
                                                       long long total) {
  long long i = (long long)blockIdx.x * blockDim.x + threadIdx.x;
  if (i >= total) return;
  int j = (int)(i % NHID);
  float v = agg[i] + b1[j];
  v = v > 0.0f ? v : 0.0f;
  v *= mask[i];
  hbuf[i] = (__bf16)v;
}

// ---------------------------------------------------------------------------
// GEMM2: sup2[N][40] (bf16) = h[N][64] (bf16) @ W2frag, N padded to 48 cols.
// ---------------------------------------------------------------------------
__global__ __launch_bounds__(128) void gemm2_wmma(const __bf16* __restrict__ hbuf,
                                                  const __bf16* __restrict__ wfrag,
                                                  __bf16* __restrict__ sup2,
                                                  int Ntiles, int N) {
  int wave = threadIdx.x >> 5;
  int lane = threadIdx.x & 31;
  int tile = blockIdx.x * 4 + wave;
  if (tile >= Ntiles) return;
  bool hi = lane >= 16;
  int m = tile * 16 + (lane & 15);
  int mc = m < N ? m : (N - 1);
  const __bf16* hrow = hbuf + (size_t)mc * NHID;
  const v16bf* bp = (const v16bf*)wfrag;

  v8f a0 = {}, a1 = {}, a2 = {};
#pragma unroll
  for (int kt = 0; kt < KT2; ++kt) {
    AFrag af;
    int c0 = kt * 32 + (hi ? 8 : 0);
    int c1 = kt * 32 + (hi ? 24 : 16);
    af.h[0] = *(const v8bf*)(hrow + c0);
    af.h[1] = *(const v8bf*)(hrow + c1);
    v16bf fb0 = bp[(kt * NT2 + 0) * 32 + lane];
    v16bf fb1 = bp[(kt * NT2 + 1) * 32 + lane];
    v16bf fb2 = bp[(kt * NT2 + 2) * 32 + lane];
    a0 = __builtin_amdgcn_wmma_f32_16x16x32_bf16(false, af.v, false, fb0, (short)0, a0, false, false);
    a1 = __builtin_amdgcn_wmma_f32_16x16x32_bf16(false, af.v, false, fb1, (short)0, a1, false, false);
    a2 = __builtin_amdgcn_wmma_f32_16x16x32_bf16(false, af.v, false, fb2, (short)0, a2, false, false);
  }
  int mbase = tile * 16 + (hi ? 8 : 0);
  int nn = lane & 15;
#pragma unroll
  for (int r = 0; r < 8; ++r) {
    int mm = mbase + r;
    if (mm < N) {
      __bf16* orow = sup2 + (size_t)mm * NCLASS;
      orow[nn] = (__bf16)a0[r];                       // cols 0..15
      orow[16 + nn] = (__bf16)a1[r];                  // cols 16..31
      if (32 + nn < NCLASS) orow[32 + nn] = (__bf16)a2[r];  // cols 32..39
    }
  }
}

// out = log_softmax(agg2 + b2) per node (row of 40)
__global__ __launch_bounds__(256) void final_logsoftmax(const float* __restrict__ agg,
                                                        const float* __restrict__ b2,
                                                        float* __restrict__ out, int N) {
  int n = blockIdx.x * blockDim.x + threadIdx.x;
  if (n >= N) return;
  const float* row = agg + (size_t)n * NCLASS;
  float mx = -3.402823466e38f;
  for (int j = 0; j < NCLASS; ++j) mx = fmaxf(mx, row[j] + b2[j]);
  float s = 0.0f;
  for (int j = 0; j < NCLASS; ++j) s += expf(row[j] + b2[j] - mx);
  float l = mx + logf(s);
  float* orow = out + (size_t)n * NCLASS;
  for (int j = 0; j < NCLASS; ++j) orow[j] = row[j] + b2[j] - l;
}

extern "C" void kernel_launch(void* const* d_in, const int* in_sizes, int n_in,
                              void* d_out, int out_size, void* d_ws, size_t ws_size,
                              hipStream_t stream) {
  const float* x    = (const float*)d_in[0];
  const float* ew   = (const float*)d_in[1];
  const float* W1   = (const float*)d_in[2];
  const float* b1   = (const float*)d_in[3];
  const float* W2   = (const float*)d_in[4];
  const float* b2   = (const float*)d_in[5];
  const float* mask = (const float*)d_in[6];
  const int*   ei   = (const int*)d_in[7];

  int N = in_sizes[0] / NFEAT;   // 100000
  int E = in_sizes[1];           // 1600000
  const int* srcv = ei;
  const int* dstv = ei + E;

  // workspace carve-out (256B aligned blocks)
  char* ws = (char*)d_ws;
  size_t off = 0;
  auto take = [&](size_t bytes) -> char* {
    char* p = ws + off;
    off += (bytes + 255) & ~(size_t)255;
    return p;
  };
  __bf16* w1f  = (__bf16*)take((size_t)KT1 * NT1 * 32 * 16 * sizeof(__bf16)); // 64 KB
  __bf16* w2f  = (__bf16*)take((size_t)KT2 * NT2 * 32 * 16 * sizeof(__bf16)); //  6 KB
  __bf16* sup1 = (__bf16*)take((size_t)N * NHID * sizeof(__bf16));            // 12.8 MB (reused as h)
  float*  agg1 = (float*) take((size_t)N * NHID * sizeof(float));             // 25.6 MB
  __bf16* sup2 = (__bf16*)take((size_t)N * NCLASS * sizeof(__bf16));          //  8.0 MB
  float*  agg2 = (float*) take((size_t)N * NCLASS * sizeof(float));           // 16.0 MB

  // 1) weight fragment prep (cheap, deterministic every call)
  {
    int t1 = KT1 * NT1 * 32 * 16;
    prep_wfrag<<<(t1 + 255) / 256, 256, 0, stream>>>(W1, w1f, NFEAT, NHID, NT1, KT1);
    int t2 = KT2 * NT2 * 32 * 16;
    prep_wfrag<<<(t2 + 255) / 256, 256, 0, stream>>>(W2, w2f, NHID, NCLASS, NT2, KT2);
  }
  // 2) zero scatter accumulators (must happen every launch)
  {
    long long n1 = (long long)N * NHID;
    zero_f32<<<(unsigned)((n1 + 255) / 256), 256, 0, stream>>>(agg1, n1);
    long long n2 = (long long)N * NCLASS;
    zero_f32<<<(unsigned)((n2 + 255) / 256), 256, 0, stream>>>(agg2, n2);
  }
  int Ntiles = (N + 15) / 16;   // 6250
  // 3) GEMM1 (WMMA bf16, fp32 accumulate)
  gemm1_wmma<<<(Ntiles + 3) / 4, 128, 0, stream>>>(x, w1f, sup1, Ntiles, N);
  // 4) edge aggregation layer 1 (64 feats, 16 chunks/edge)
  {
    long long tt = (long long)E * (NHID / 4);
    aggregate_bf16<<<(unsigned)((tt + 255) / 256), 256, 0, stream>>>(
        srcv, dstv, ew, sup1, agg1, E, NHID, NHID / 4);
  }
  // 5) bias + relu + dropout -> h (bf16, overwrite sup1)
  {
    long long tt = (long long)N * NHID;
    post1_relu_drop<<<(unsigned)((tt + 255) / 256), 256, 0, stream>>>(agg1, b1, mask, sup1, tt);
  }
  // 6) GEMM2 (WMMA bf16)
  gemm2_wmma<<<(Ntiles + 3) / 4, 128, 0, stream>>>(sup1, w2f, sup2, Ntiles, N);
  // 7) edge aggregation layer 2 (40 feats, 10 chunks/edge)
  {
    long long tt = (long long)E * (NCLASS / 4);
    aggregate_bf16<<<(unsigned)((tt + 255) / 256), 256, 0, stream>>>(
        srcv, dstv, ew, sup2, agg2, E, NCLASS, NCLASS / 4);
  }
  // 8) bias + log_softmax -> d_out
  final_logsoftmax<<<(N + 255) / 256, 256, 0, stream>>>(agg2, b2, (float*)d_out, N);
}
